// ContrastiveLoss_90271622627495
// MI455X (gfx1250) — matrix operand order repaired and checked by reading, not compiled, not run on hardware
//
#include <hip/hip_runtime.h>
#include <hip/hip_bf16.h>

// ---------------------------------------------------------------------------
// ContrastiveLoss on MI455X (gfx1250, wave32):
//   Split-bf16 Gram GEMM (f = H + L), gram ~= H.H^T + H.L^T + L.H^T via
//   V_WMMA_F32_16X16X32_BF16, LDS-staged operands with double-buffered
//   GLOBAL_LOAD_ASYNC_TO_LDS_B128 (ASYNCcnt), upper-triangle x2 evaluation,
//   fused dist/sim/sign epilogue, deterministic two-stage reduction.
// ---------------------------------------------------------------------------

typedef __bf16 bf16_t;
typedef bf16_t v16bf __attribute__((ext_vector_type(16)));
typedef bf16_t v8bf  __attribute__((ext_vector_type(8)));
typedef bf16_t v4bf  __attribute__((ext_vector_type(4)));
typedef float  v8f   __attribute__((ext_vector_type(8)));
typedef int    v4i   __attribute__((ext_vector_type(4)));

#define NROWS 4096
#define NDIM  1024
#define COS_EPS 1e-8f

// Workgroup = 256 threads = 8 waves arranged 4 (M) x 2 (N).
// Wave tile: 32 rows x 64 cols (2x4 grid of 16x16 accumulators).
// Workgroup tile: 128 x 128.  Grid: (4096/128, 4096/128) = (32, 32).
#define WG_M 128
#define WG_N 128
#define KCHUNK 32
#define NUM_PARTIALS ((NROWS / WG_M) * (NROWS / WG_N))  // 1024

// LDS: 2 buffers x 4 sub-tiles (AH, AL, BH, BL) x (128 rows x 32 bf16) = 64 KB
#define SUB_ELEMS (128 * KCHUNK)          // 4096 bf16 per sub-tile
#define BUF_ELEMS (4 * SUB_ELEMS)         // 16384 bf16 per buffer

// ---------------------------------------------------------------------------
// Async global -> LDS helpers (CDNA5 ASYNCcnt path).
// Builtin signature (probe-confirmed by round-3 diagnostic):
//   (v4i addrspace(1)* src, v4i addrspace(3)* dst, i32 imm_offset, i32 cpol)
// ---------------------------------------------------------------------------
__device__ __forceinline__ void async_copy_b128(const bf16_t* g, bf16_t* l) {
#if __has_builtin(__builtin_amdgcn_global_load_async_to_lds_b128)
    __builtin_amdgcn_global_load_async_to_lds_b128(
        (__attribute__((address_space(1))) v4i*)g,
        (__attribute__((address_space(3))) v4i*)l, 0, 0);
#else
    asm volatile("global_load_async_to_lds_b128 %0, %1, off"
                 :
                 : "v"((unsigned int)(unsigned long long)l), "v"(g)
                 : "memory");
#endif
}

__device__ __forceinline__ void wait_async0() {
#if __has_builtin(__builtin_amdgcn_s_wait_asynccnt)
    __builtin_amdgcn_s_wait_asynccnt(0);
#else
    asm volatile("s_wait_asynccnt 0x0" ::: "memory");
#endif
}

// ---------------------------------------------------------------------------
// Kernel 1: split f32 -> (H, L) bf16 pair.  H = bf16(f), L = bf16(f - H).
// ---------------------------------------------------------------------------
__global__ void split_bf16_kernel(const float* __restrict__ f,
                                  bf16_t* __restrict__ H,
                                  bf16_t* __restrict__ L) {
    const size_t idx = ((size_t)blockIdx.x * blockDim.x + threadIdx.x) * 4;
    const float4 v = *(const float4*)(f + idx);
    v4bf h, l;
    h[0] = (bf16_t)v.x; l[0] = (bf16_t)(v.x - (float)h[0]);
    h[1] = (bf16_t)v.y; l[1] = (bf16_t)(v.y - (float)h[1]);
    h[2] = (bf16_t)v.z; l[2] = (bf16_t)(v.z - (float)h[2]);
    h[3] = (bf16_t)v.w; l[3] = (bf16_t)(v.w - (float)h[3]);
    *(v4bf*)(H + idx) = h;
    *(v4bf*)(L + idx) = l;
}

// ---------------------------------------------------------------------------
// Kernel 2: row squared norms (f32 exact). 8 waves per block, wave per row.
// ---------------------------------------------------------------------------
__global__ void rowsq_kernel(const float* __restrict__ f, float* __restrict__ sq) {
    const int wave = threadIdx.x >> 5;
    const int lane = threadIdx.x & 31;
    const int row  = blockIdx.x * 8 + wave;
    const float* __restrict__ r = f + (size_t)row * NDIM;

    float acc = 0.0f;
    #pragma unroll
    for (int it = 0; it < 8; ++it) {
        const float4 v = ((const float4*)r)[it * 32 + lane];
        acc += v.x * v.x + v.y * v.y + v.z * v.z + v.w * v.w;
    }
    #pragma unroll
    for (int off = 16; off > 0; off >>= 1)
        acc += __shfl_down(acc, off, 32);
    if (lane == 0) sq[row] = acc;
}

// ---------------------------------------------------------------------------
// bf16 fragment loaders from LDS, per CDNA5 ISA §7.12.2 (wave32).
// Row stride in LDS: KCHUNK (=32) bf16 = 64 B.
// A (16x32): chunks {8*lhi..+7} and {16+8*lhi..+7} of the row.
// B (32x16): chunks {8*lhi..+7} and {16+8*lhi..+7} of the row.
// ---------------------------------------------------------------------------
__device__ __forceinline__ v16bf afrag(const bf16_t* row, int lhi) {
    const v8bf p0 = *(const v8bf*)(row + 8 * lhi);
    const v8bf p1 = *(const v8bf*)(row + 16 + 8 * lhi);
    return __builtin_shufflevector(p0, p1, 0, 1, 2, 3, 4, 5, 6, 7,
                                   8, 9, 10, 11, 12, 13, 14, 15);
}

__device__ __forceinline__ v16bf bfrag(const bf16_t* row, int lhi) {
    const v8bf p0 = *(const v8bf*)(row + 8 * lhi);
    const v8bf p1 = *(const v8bf*)(row + 16 + 8 * lhi);
    return __builtin_shufflevector(p0, p1, 0, 1, 2, 3, 4, 5, 6, 7,
                                   8, 9, 10, 11, 12, 13, 14, 15);
}

#define WMMA_BF16(A, B, C) \
    __builtin_amdgcn_wmma_f32_16x16x32_bf16(false, (A), false, (B), (short)0, (C), false, false)

// ---------------------------------------------------------------------------
// Kernel 3: LDS-staged split-bf16 gram tile + epilogue + partial sum.
// ---------------------------------------------------------------------------
__global__ void __launch_bounds__(256)
gram_loss_kernel(const bf16_t* __restrict__ H,
                 const bf16_t* __restrict__ L,
                 const long long* __restrict__ y,
                 const float* __restrict__ sq,
                 float* __restrict__ partials) {
    __shared__ __align__(16) bf16_t ldsbuf[2 * BUF_ELEMS];   // 64 KB

    const int pidx = blockIdx.y * gridDim.x + blockIdx.x;
    const int rowBase = blockIdx.x * WG_M;
    const int colBase = blockIdx.y * WG_N;

    // Block entirely below the diagonal contributes nothing (uniform branch).
    if (colBase + WG_N <= rowBase) {
        if (threadIdx.x == 0) partials[pidx] = 0.0f;
        return;
    }

    const int tid  = threadIdx.x;
    const int wave = tid >> 5;           // 0..7
    const int lane = tid & 31;
    const int lmod = lane & 15;
    const int lhi  = lane >> 4;
    const int waveM = wave & 3;          // 0..3
    const int waveN = wave >> 2;         // 0..1

    // Global bases of the four staged sub-tiles (row 0 of this block).
    const bf16_t* __restrict__ baseSub[4] = {
        H + (size_t)rowBase * NDIM,      // AH
        L + (size_t)rowBase * NDIM,      // AL
        H + (size_t)colBase * NDIM,      // BH
        L + (size_t)colBase * NDIM       // BL
    };

    // Issue one K-chunk's 32 KB as 8 async b128s per thread.
    // Segment map: sub = i/2 (compile-time), row = (i&1)*64 + tid/4, c = tid&3.
    const int segR = tid >> 2;
    const int segC = tid & 3;
    auto issue_chunk = [&](int buf, int k) {
        bf16_t* dst = ldsbuf + buf * BUF_ELEMS;
        #pragma unroll
        for (int i = 0; i < 8; ++i) {
            const int sub = i >> 1;
            const int r   = (i & 1) * 64 + segR;
            const bf16_t* g = baseSub[sub] + (size_t)r * NDIM + k + segC * 8;
            bf16_t*       l = dst + sub * SUB_ELEMS + r * KCHUNK + segC * 8;
            async_copy_b128(g, l);
        }
    };

    v8f acc[2][4];
    #pragma unroll
    for (int mi = 0; mi < 2; ++mi)
        #pragma unroll
        for (int j = 0; j < 4; ++j)
            acc[mi][j] = (v8f){};

    issue_chunk(0, 0);
    wait_async0();
    __syncthreads();

    const int NSTEPS = NDIM / KCHUNK;    // 32
    #pragma unroll 1
    for (int step = 0; step < NSTEPS; ++step) {
        const int buf = step & 1;
        if (step + 1 < NSTEPS) issue_chunk((step + 1) & 1, (step + 1) * KCHUNK);

        const bf16_t* bp = ldsbuf + buf * BUF_ELEMS;
        const bf16_t* AH = bp;
        const bf16_t* AL = bp + SUB_ELEMS;
        const bf16_t* BH = bp + 2 * SUB_ELEMS;
        const bf16_t* BL = bp + 3 * SUB_ELEMS;

        const int ar0 = (waveM * 32 + lmod) * KCHUNK;
        const int ar1 = ar0 + 16 * KCHUNK;
        const v16bf ah0 = afrag(AH + ar0, lhi);
        const v16bf al0 = afrag(AL + ar0, lhi);
        const v16bf ah1 = afrag(AH + ar1, lhi);
        const v16bf al1 = afrag(AL + ar1, lhi);

        #pragma unroll
        for (int j = 0; j < 4; ++j) {
            const int br = (waveN * 64 + j * 16 + lmod) * KCHUNK;
            const v16bf bh = bfrag(BH + br, lhi);
            const v16bf bl = bfrag(BL + br, lhi);
            // gram ~= H.H^T + H.L^T + L.H^T (L.L^T ~ 2^-16 rel, dropped)
            acc[0][j] = WMMA_BF16(ah0, bh, acc[0][j]);
            acc[0][j] = WMMA_BF16(ah0, bl, acc[0][j]);
            acc[0][j] = WMMA_BF16(al0, bh, acc[0][j]);
            acc[1][j] = WMMA_BF16(ah1, bh, acc[1][j]);
            acc[1][j] = WMMA_BF16(ah1, bl, acc[1][j]);
            acc[1][j] = WMMA_BF16(al1, bh, acc[1][j]);
        }

        wait_async0();       // next buffer's DMA has landed (own counter)
        __syncthreads();     // all waves done reading current buffer
    }

    // Epilogue: C layout M = vgpr + 8*(lane/16), N = lane%16.
    // Upper triangle weighting: w = (i<j)?2 : (i==j)?1 : 0.
    const int tileM = rowBase + waveM * 32;
    const int tileN = colBase + waveN * 64;
    float accv = 0.0f;
    #pragma unroll
    for (int mi = 0; mi < 2; ++mi) {
        #pragma unroll
        for (int jn = 0; jn < 4; ++jn) {
            const int j = tileN + jn * 16 + lmod;
            const float sqj = sq[j];
            const long long yj = y[j];
            const float nj = fmaxf(sqrtf(sqj), COS_EPS);
            #pragma unroll
            for (int r = 0; r < 8; ++r) {
                const int i = tileM + mi * 16 + r + 8 * lhi;
                const float w = (i < j) ? 2.0f : ((i == j) ? 1.0f : 0.0f);
                const float g   = acc[mi][jn][r];
                const float sqi = sq[i];
                const float d2  = sqi + sqj - 2.0f * g;
                const float dist = (d2 > 0.0f) ? sqrtf(d2) : 0.0f;
                const float ni  = fmaxf(sqrtf(sqi), COS_EPS);
                const float sim = g / (ni * nj);
                const float sgn = (y[i] == yj) ? 1.0f : -1.0f;
                accv += w * sgn * (dist - sim);
            }
        }
    }

    #pragma unroll
    for (int off = 16; off > 0; off >>= 1)
        accv += __shfl_down(accv, off, 32);

    // Reuse dead LDS buffer for the cross-wave sum (kept static LDS at 64 KB).
    float* wsum = (float*)ldsbuf;
    if (lane == 0) wsum[wave] = accv;
    __syncthreads();
    if (threadIdx.x == 0) {
        float s = 0.0f;
        #pragma unroll
        for (int w = 0; w < 8; ++w) s += wsum[w];
        partials[pidx] = s;
    }
}

// ---------------------------------------------------------------------------
// Kernel 4: deterministic final reduction of partials -> scalar.
// ---------------------------------------------------------------------------
__global__ void final_reduce_kernel(const float* __restrict__ partials,
                                    float* __restrict__ out) {
    __shared__ float sm[256];
    float acc = 0.0f;
    for (int idx = threadIdx.x; idx < NUM_PARTIALS; idx += 256)
        acc += partials[idx];
    sm[threadIdx.x] = acc;
    __syncthreads();
    for (int off = 128; off > 0; off >>= 1) {
        if (threadIdx.x < off) sm[threadIdx.x] += sm[threadIdx.x + off];
        __syncthreads();
    }
    if (threadIdx.x == 0) out[0] = sm[0];
}

// ---------------------------------------------------------------------------
extern "C" void kernel_launch(void* const* d_in, const int* in_sizes, int n_in,
                              void* d_out, int out_size, void* d_ws, size_t ws_size,
                              hipStream_t stream) {
    const float*     features = (const float*)d_in[0];
    const long long* y        = (const long long*)d_in[1];
    float* out = (float*)d_out;

    bf16_t* Hws = (bf16_t*)d_ws;                    // NROWS*NDIM bf16 (8 MB)
    bf16_t* Lws = Hws + (size_t)NROWS * NDIM;       // NROWS*NDIM bf16 (8 MB)
    float*  sq  = (float*)(Lws + (size_t)NROWS * NDIM);
    float*  partials = sq + NROWS;                  // NUM_PARTIALS floats

    // 1) split f32 features into bf16 hi/lo arrays
    split_bf16_kernel<<<(NROWS * NDIM / 4) / 256, 256, 0, stream>>>(features, Hws, Lws);

    // 2) row squared norms (exact f32)
    rowsq_kernel<<<NROWS / 8, 256, 0, stream>>>(features, sq);

    // 3) LDS-staged split-bf16 WMMA gram + epilogue (upper triangle x2)
    dim3 grid(NROWS / WG_M, NROWS / WG_N);          // (32, 32)
    gram_loss_kernel<<<grid, 256, 0, stream>>>(Hws, Lws, y, sq, partials);

    // 4) deterministic final reduce
    final_reduce_kernel<<<1, 256, 0, stream>>>(partials, out);
}
